// GateDeltaNet_40673340293279
// MI455X (gfx1250) — matrix operand order repaired and checked by reading, not compiled, not run on hardware
//
#include <hip/hip_runtime.h>

// ---------------- problem constants ----------------
#define SEQ    8192
#define HID    2048
#define HKH    8
#define HVH    16
#define DKD    128
#define DVD    128
#define KCONV  4
#define CHUNK  64
#define NCH    (SEQ / CHUNK)               // 128
#define CONVD  (2 * HKH * DKD + HVH * DVD) // 4096
#define EPSV   1e-6f

typedef __attribute__((ext_vector_type(16))) _Float16 v16h;
typedef __attribute__((ext_vector_type(8)))  _Float16 h8;
typedef __attribute__((ext_vector_type(8)))  float    v8f;

#define DEVINL __device__ __forceinline__

DEVINL float sigmoidf_(float x) { return 1.0f / (1.0f + __expf(-x)); }
DEVINL float siluf_(float x)    { return x * sigmoidf_(x); }
DEVINL float softplusf_(float x){ return (x > 20.0f) ? x : log1pf(__expf(x)); }

DEVINL v8f wmma16(v16h a, v16h b, v8f c) {
  return __builtin_amdgcn_wmma_f32_16x16x32_f16(false, a, false, b, (short)0, c, false, false);
}

DEVINL h8 ldh8(const _Float16* p) { return *(const h8*)p; }

DEVINL v16h mkfrag(h8 lo, h8 hi) {
  return __builtin_shufflevector(lo, hi, 0, 1, 2, 3, 4, 5, 6, 7, 8, 9, 10, 11, 12, 13, 14, 15);
}

DEVINL v16h scale16(v16h v, float s) {
  v16h r;
  #pragma unroll
  for (int e = 0; e < 16; ++e) r[e] = (_Float16)((float)v[e] * s);
  return r;
}

DEVINL float wavesum(float v) {
  #pragma unroll
  for (int off = 16; off > 0; off >>= 1) v += __shfl_xor(v, off, 32);
  return v;
}

// ---- CDNA5 async global->LDS copy (ASYNCcnt-tracked, bypasses VGPRs) ----
DEVINL void async_copy_b128(void* lds, const void* glob) {
  unsigned loff = (unsigned)(uintptr_t)(__attribute__((address_space(3))) void*)lds;
  asm volatile("global_load_async_to_lds_b128 %0, %1, off"
               :: "v"(loff), "v"(glob) : "memory");
}
DEVINL void wait_asynccnt0() {
  asm volatile("s_wait_asynccnt 0x0" ::: "memory");
}

// ---------------- f32 -> f16 convert (8 elems/thread, b128 traffic) ----------------
__global__ void gdn_cvt_f16(const float* __restrict__ s, _Float16* __restrict__ d, size_t n) {
  size_t i = ((size_t)blockIdx.x * 256 + threadIdx.x) * 8;
  if (i + 8 <= n) {
    float4 a = *(const float4*)(s + i);
    float4 b = *(const float4*)(s + i + 4);
    h8 o;
    o[0] = (_Float16)a.x; o[1] = (_Float16)a.y; o[2] = (_Float16)a.z; o[3] = (_Float16)a.w;
    o[4] = (_Float16)b.x; o[5] = (_Float16)b.y; o[6] = (_Float16)b.z; o[7] = (_Float16)b.w;
    *(h8*)(d + i) = o;
  }
}

// ---------------- WMMA GEMM: C[M,N] = A[M,K] @ B[K,N], f16 in f32 out -----------
// block tile 128x64, 8 waves each owning 32x32 (2x2 wmma accumulators)
// A tile staged with async global->LDS b128; B tile loaded b128 + transposed scatter.
#define GBM 128
#define GBN 64
#define GBK 32
#define ALD (GBK + 8)
__global__ __launch_bounds__(256) void gdn_gemm_wmma(
    const _Float16* __restrict__ A, const _Float16* __restrict__ B,
    float* __restrict__ C, int M, int N, int K)
{
  __shared__ _Float16 As[GBM][ALD];    // row-major A tile
  __shared__ _Float16 BsT[GBN][ALD];   // transposed B tile: [n][k]
  const int tid = threadIdx.x, lane = tid & 31, wave = tid >> 5;
  const int m_ = lane & 15, grp = lane >> 4;
  const int bm = blockIdx.y * GBM, bn = blockIdx.x * GBN;
  const int wr = (wave >> 1) * 32, wc = (wave & 1) * 32;
  v8f acc[2][2] = {};
  for (int k0 = 0; k0 < K; k0 += GBK) {
    // stage A: 512 groups of 8 contiguous halfs, async DMA into LDS
    for (int g = tid; g < (GBM * GBK) / 8; g += 256) {
      int r = g >> 2, c8 = (g & 3) * 8;
      async_copy_b128(&As[r][c8], &A[(size_t)(bm + r) * K + k0 + c8]);
    }
    // stage B transposed: 256 groups (one per thread)
    {
      int kr = tid >> 3, n0 = (tid & 7) * 8;
      h8 bv = ldh8(&B[(size_t)(k0 + kr) * N + bn + n0]);
      #pragma unroll
      for (int i = 0; i < 8; ++i) BsT[n0 + i][kr] = bv[i];
    }
    wait_asynccnt0();
    __syncthreads();
    v16h af[2], bf[2];
    #pragma unroll
    for (int mt = 0; mt < 2; ++mt) {
      int row = wr + mt * 16 + m_;
      af[mt] = mkfrag(ldh8(&As[row][grp * 8]), ldh8(&As[row][16 + grp * 8]));
    }
    #pragma unroll
    for (int nt = 0; nt < 2; ++nt) {
      int col = wc + nt * 16 + m_;
      bf[nt] = mkfrag(ldh8(&BsT[col][grp * 8]), ldh8(&BsT[col][16 + grp * 8]));
    }
    #pragma unroll
    for (int mt = 0; mt < 2; ++mt)
      #pragma unroll
      for (int nt = 0; nt < 2; ++nt)
        acc[mt][nt] = wmma16(af[mt], bf[nt], acc[mt][nt]);
    __syncthreads();
  }
  #pragma unroll
  for (int mt = 0; mt < 2; ++mt)
    #pragma unroll
    for (int nt = 0; nt < 2; ++nt)
      #pragma unroll
      for (int r = 0; r < 8; ++r) {
        int gr = bm + wr + mt * 16 + grp * 8 + r;
        int gc = bn + wc + nt * 16 + m_;
        C[(size_t)gr * N + gc] = acc[mt][nt][r];
      }
}

// ---------------- small projections b = x@Wb, a = x@Wa (N=16 each) ----------------
__global__ __launch_bounds__(256) void gdn_proj_ba(
    const float* __restrict__ x, const float* __restrict__ Wb, const float* __restrict__ Wa,
    float* __restrict__ bvec, float* __restrict__ avec)
{
  int s = blockIdx.x * 8 + (threadIdx.x >> 5);
  int lane = threadIdx.x & 31;
  if (s >= SEQ) return;
  float accb[16], acca[16];
  #pragma unroll
  for (int o = 0; o < 16; ++o) { accb[o] = 0.f; acca[o] = 0.f; }
  for (int k = lane; k < HID; k += 32) {
    float xv = x[(size_t)s * HID + k];
    #pragma unroll
    for (int o = 0; o < 16; ++o) {
      accb[o] += xv * Wb[k * HVH + o];
      acca[o] += xv * Wa[k * HVH + o];
    }
  }
  #pragma unroll
  for (int o = 0; o < 16; ++o) {
    float b = wavesum(accb[o]);
    float a = wavesum(acca[o]);
    if (lane == 0) {
      bvec[(size_t)s * HVH + o] = b;
      avec[(size_t)s * HVH + o] = a;
    }
  }
}

// ---------------- causal depthwise conv (K=4) + silu + split/route ----------------
__global__ void gdn_conv_silu(const float* __restrict__ mixed, const float* __restrict__ convw,
                              float* __restrict__ qraw, float* __restrict__ kraw,
                              _Float16* __restrict__ vh)
{
  size_t idx = (size_t)blockIdx.x * 256 + threadIdx.x;
  if (idx >= (size_t)SEQ * CONVD) return;
  int ch = (int)(idx % CONVD);
  int s  = (int)(idx / CONVD);
  float acc = 0.0f;
  #pragma unroll
  for (int j = 0; j < KCONV; ++j) {
    int sp = s - (KCONV - 1) + j;
    if (sp >= 0) acc += mixed[(size_t)sp * CONVD + ch] * convw[ch * KCONV + j];
  }
  float y = siluf_(acc);
  if (ch < HKH * DKD)            qraw[(size_t)s * HKH * DKD + ch] = y;
  else if (ch < 2 * HKH * DKD)   kraw[(size_t)s * HKH * DKD + (ch - HKH * DKD)] = y;
  else                           vh[(size_t)s * HVH * DVD + (ch - 2 * HKH * DKD)] = (_Float16)y;
}

// ---------------- L2 norm of q (scaled by DK^-0.5) and k, write f16 ----------------
__global__ __launch_bounds__(256) void gdn_l2norm_qk(
    const float* __restrict__ qraw, const float* __restrict__ kraw,
    _Float16* __restrict__ qn, _Float16* __restrict__ kn)
{
  int wid = blockIdx.x * 8 + (threadIdx.x >> 5);
  int lane = threadIdx.x & 31;
  const int total = 2 * SEQ * HKH;
  if (wid >= total) return;
  bool isq = wid < SEQ * HKH;
  int row = isq ? wid : (wid - SEQ * HKH);
  const float* src = (isq ? qraw : kraw) + (size_t)row * DKD;
  float v[4]; float ss = 0.0f;
  #pragma unroll
  for (int i = 0; i < 4; ++i) { v[i] = src[lane + 32 * i]; ss += v[i] * v[i]; }
  ss = wavesum(ss);
  float r = rsqrtf(ss + EPSV);
  if (isq) r *= rsqrtf((float)DKD);
  _Float16* dst = (isq ? qn : kn) + (size_t)row * DKD;
  #pragma unroll
  for (int i = 0; i < 4; ++i) dst[lane + 32 * i] = (_Float16)(v[i] * r);
}

// ---------------- beta = sigmoid(b), g = -exp(A_log)*softplus(a + dt_bias) --------
__global__ void gdn_gates(const float* __restrict__ bvec, const float* __restrict__ avec,
                          const float* __restrict__ Alog, const float* __restrict__ dtb,
                          float* __restrict__ beta, float* __restrict__ g)
{
  int i = blockIdx.x * 256 + threadIdx.x;
  if (i >= SEQ * HVH) return;
  int h = i % HVH;
  beta[i] = sigmoidf_(bvec[i]);
  g[i] = -__expf(Alog[h]) * softplusf_(avec[i] + dtb[h]);
}

// ---------------- per-chunk cumulative sum of g ----------------
__global__ void gdn_cumsum(const float* __restrict__ g, float* __restrict__ gc)
{
  int i = blockIdx.x * 256 + threadIdx.x;
  if (i >= NCH * HVH) return;
  int h = i % HVH, n = i / HVH;
  float acc = 0.0f;
  for (int c = 0; c < CHUNK; ++c) {
    acc += g[(size_t)(n * CHUNK + c) * HVH + h];
    gc[(size_t)(n * CHUNK + c) * HVH + h] = acc;
  }
}

// ---------------- per (chunk, head): A, T=(I-A)^-1, w=T@(beta v), u=T@(k_beta e^g), Aqk
// dynamic LDS layout (bytes):
//   kL  [64][136] f16 @ 0       (17408)   k rows (contiguous along d)
//   T16 [64][72]  f16 @ 17408   ( 9216)   f16 mirror of T
//   vT  [128][72] f16 @ 26624   (18432)   (beta*v) transposed: [j][c]
//   kuT [128][72] f16 @ 45056   (18432)   (beta*e^g*k) transposed: [d][c]
//   AL  [64][68]  f32 @ 63488   (17408)
//   TL  [64][68]  f32 @ 80896   (17408)
//   betaL[64]     f32 @ 98304, gcL[64] f32 @ 98560   -> total 98816
#define CHK_LDS 98816
__global__ __launch_bounds__(256) void gdn_chunk(
    const _Float16* __restrict__ qn, const _Float16* __restrict__ kn,
    const _Float16* __restrict__ vh, const float* __restrict__ beta,
    const float* __restrict__ gc,
    _Float16* __restrict__ w16, _Float16* __restrict__ u16, _Float16* __restrict__ aqk16)
{
  extern __shared__ unsigned char cs[];
  _Float16* kL   = (_Float16*)cs;               // stride 136
  _Float16* T16  = (_Float16*)(cs + 17408);     // stride 72
  _Float16* vT   = (_Float16*)(cs + 26624);     // stride 72
  _Float16* kuT  = (_Float16*)(cs + 45056);     // stride 72
  float*    AL   = (float*)(cs + 63488);        // stride 68
  float*    TL   = (float*)(cs + 80896);        // stride 68
  float*    betaL= (float*)(cs + 98304);
  float*    gcL  = (float*)(cs + 98560);

  const int n = blockIdx.x, hv = blockIdx.y;
  const int hk = hv >> 1, hq = hv >> 1;
  const int s0 = n * CHUNK;
  const int tid = threadIdx.x, lane = tid & 31, wave = tid >> 5;
  const int m_ = lane & 15, grp = lane >> 4;

  for (int g = tid; g < CHUNK * DKD / 8; g += 256) {
    int c = g >> 4, d0 = (g & 15) * 8;
    *(h8*)&kL[c * 136 + d0] = ldh8(&kn[((size_t)(s0 + c) * HKH + hk) * DKD + d0]);
  }
  if (tid < CHUNK) {
    betaL[tid] = beta[(size_t)(s0 + tid) * HVH + hv];
    gcL[tid]   = gc[(size_t)(s0 + tid) * HVH + hv];
  }
  __syncthreads();

  // build transposed f16 operands: kuT[d][c] = k*beta*e^g, vT[j][c] = v*beta
  for (int g = tid; g < CHUNK * DKD / 8; g += 256) {
    int c = g >> 4, d0 = (g & 15) * 8;
    float sc = betaL[c] * __expf(gcL[c]);
    h8 kv = *(const h8*)&kL[c * 136 + d0];
    #pragma unroll
    for (int i = 0; i < 8; ++i) kuT[(d0 + i) * 72 + c] = (_Float16)((float)kv[i] * sc);
  }
  for (int g = tid; g < CHUNK * DVD / 8; g += 256) {
    int c = g >> 4, j0 = (g & 15) * 8;
    float be = betaL[c];
    h8 vv = ldh8(&vh[((size_t)(s0 + c) * HVH + hv) * DVD + j0]);
    #pragma unroll
    for (int i = 0; i < 8; ++i) vT[(j0 + i) * 72 + c] = (_Float16)((float)vv[i] * be);
  }
  __syncthreads();

  // A_raw = k_beta @ k^T (64x64, K=128): 16 tiles
  for (int t = wave; t < 16; t += 8) {
    int tm = t >> 2, tn = t & 3;
    v8f acc = {};
    #pragma unroll
    for (int kk = 0; kk < DKD; kk += 32) {
      int row = tm * 16 + m_;
      v16h af = scale16(mkfrag(ldh8(&kL[row * 136 + kk + grp * 8]),
                               ldh8(&kL[row * 136 + kk + 16 + grp * 8])), betaL[row]);
      v16h bf = mkfrag(ldh8(&kL[(tn * 16 + m_) * 136 + kk + grp * 8]),
                       ldh8(&kL[(tn * 16 + m_) * 136 + kk + 16 + grp * 8]));
      acc = wmma16(af, bf, acc);
    }
    #pragma unroll
    for (int r = 0; r < 8; ++r) {
      int i_ = tm * 16 + grp * 8 + r, j_ = tn * 16 + m_;
      AL[i_ * 68 + j_] = (i_ > j_) ? (-acc[r] * __expf(gcL[i_] - gcL[j_])) : 0.0f;
      TL[i_ * 68 + j_] = (i_ == j_) ? 1.0f : 0.0f;
    }
  }
  __syncthreads();

  // forward substitution: T[i] = e_i + sum_{k<i} A[i][k] T[k]
  for (int i_ = 1; i_ < CHUNK; ++i_) {
    if (tid < CHUNK) {
      float sum = 0.0f;
      for (int k_ = 0; k_ < i_; ++k_) sum += AL[i_ * 68 + k_] * TL[k_ * 68 + tid];
      TL[i_ * 68 + tid] += sum;
    }
    __syncthreads();
  }
  for (int i = tid; i < CHUNK * CHUNK; i += 256) {
    int r = i >> 6, c = i & 63;
    T16[r * 72 + c] = (_Float16)TL[r * 68 + c];
  }
  __syncthreads();

  // w = T @ (beta*v) and u = T @ (k_beta*e^g)  (64x128 each): 32+32 tiles
  for (int t = wave; t < 64; t += 8) {
    int which = t >> 5;                 // 0 -> w, 1 -> u
    int tt = t & 31;
    int tm = tt >> 3, tn = tt & 7;
    const _Float16* Bt = which ? kuT : vT;
    v8f acc = {};
    #pragma unroll
    for (int kk = 0; kk < CHUNK; kk += 32) {
      int row = tm * 16 + m_;
      v16h af = mkfrag(ldh8(&T16[row * 72 + kk + grp * 8]),
                       ldh8(&T16[row * 72 + kk + 16 + grp * 8]));
      int col = tn * 16 + m_;
      v16h bf = mkfrag(ldh8(&Bt[col * 72 + kk + grp * 8]),
                       ldh8(&Bt[col * 72 + kk + 16 + grp * 8]));
      acc = wmma16(af, bf, acc);
    }
    _Float16* dst = which ? u16 : w16;
    #pragma unroll
    for (int r = 0; r < 8; ++r) {
      int i_ = tm * 16 + grp * 8 + r, j_ = tn * 16 + m_;
      dst[((size_t)(s0 + i_) * HVH + hv) * DKD + j_] = (_Float16)acc[r];
    }
  }

  // Aqk = (q @ k^T) * exp(diff), incl-diagonal mask (64x64, K=128): 16 tiles
  for (int t = wave; t < 16; t += 8) {
    int tm = t >> 2, tn = t & 3;
    v8f acc = {};
    #pragma unroll
    for (int kk = 0; kk < DKD; kk += 32) {
      int row = tm * 16 + m_;
      const _Float16* qrow = qn + ((size_t)(s0 + row) * HKH + hq) * DKD;
      v16h af = mkfrag(ldh8(&qrow[kk + grp * 8]), ldh8(&qrow[kk + 16 + grp * 8]));
      v16h bf = mkfrag(ldh8(&kL[(tn * 16 + m_) * 136 + kk + grp * 8]),
                       ldh8(&kL[(tn * 16 + m_) * 136 + kk + 16 + grp * 8]));
      acc = wmma16(af, bf, acc);
    }
    #pragma unroll
    for (int r = 0; r < 8; ++r) {
      int i_ = tm * 16 + grp * 8 + r, j_ = tn * 16 + m_;
      float v = (i_ >= j_) ? (acc[r] * __expf(gcL[i_] - gcL[j_])) : 0.0f;
      aqk16[(((size_t)n * HVH + hv) * CHUNK + i_) * CHUNK + j_] = (_Float16)v;
    }
  }
}

// ---------------- sequential inter-chunk scan, one block per value head ----------
// dynamic LDS (bytes):
//   state  [128][132] f32 @ 0        (67584)  exact f32 state
//   st16T  [128][136] f16 @ 67584    (34816)  f16 state transposed: [v][d]
//   vnT    [128][72]  f16 @ 102400   (18432)  v_new transposed: [j][c]
//   kdT    [128][72]  f16 @ 120832   (18432)  k_dec transposed: [d][c]
//   gcL    [64] f32  @ 139264        -> total 139520
#define SCAN_LDS 139520
__global__ __launch_bounds__(256) void gdn_scan(
    const _Float16* __restrict__ qn, const _Float16* __restrict__ kn,
    const _Float16* __restrict__ w16, const _Float16* __restrict__ u16,
    const _Float16* __restrict__ aqk16, const float* __restrict__ gc,
    float* __restrict__ core)
{
  extern __shared__ unsigned char smem[];
  float*    state = (float*)smem;                  // stride 132
  _Float16* st16T = (_Float16*)(smem + 67584);     // stride 136
  _Float16* vnT   = (_Float16*)(smem + 102400);    // stride 72
  _Float16* kdT   = (_Float16*)(smem + 120832);    // stride 72
  float*    gcL   = (float*)(smem + 139264);

  const int hv = blockIdx.x;
  const int hk = hv >> 1, hq = hv >> 1;
  const int tid = threadIdx.x, lane = tid & 31, wave = tid >> 5;
  const int m_ = lane & 15, grp = lane >> 4;

  for (int i = tid; i < DKD * 132; i += 256) state[i] = 0.0f;
  for (int i = tid; i < DKD * 136; i += 256) st16T[i] = (_Float16)0.0f;
  __syncthreads();

  for (int n = 0; n < NCH; ++n) {
    const int s0 = n * CHUNK;
    if (tid < CHUNK) gcL[tid] = gc[(size_t)(s0 + tid) * HVH + hv];
    __syncthreads();
    const float glast = gcL[CHUNK - 1];
    const float el = __expf(glast);

    // stage kdT[d][c] = k[c][d] * exp(glast - gc[c])
    for (int g = tid; g < CHUNK * DKD / 8; g += 256) {
      int c = g >> 4, d0 = (g & 15) * 8;
      float sc = __expf(glast - gcL[c]);
      h8 kv = ldh8(&kn[((size_t)(s0 + c) * HKH + hk) * DKD + d0]);
      #pragma unroll
      for (int i = 0; i < 8; ++i) kdT[(d0 + i) * 72 + c] = (_Float16)((float)kv[i] * sc);
    }
    __syncthreads();

    // P1: vnew = w - u @ state   (64x128, K=128)
    for (int t = wave; t < 32; t += 8) {
      int tm = t >> 3, tn = t & 7;
      v8f acc = {};
      #pragma unroll
      for (int kk = 0; kk < DKD; kk += 32) {
        const _Float16* urow = u16 + ((size_t)(s0 + tm * 16 + m_) * HVH + hv) * DKD;
        v16h af = mkfrag(ldh8(&urow[kk + grp * 8]), ldh8(&urow[kk + 16 + grp * 8]));
        int col = tn * 16 + m_;
        v16h bf = mkfrag(ldh8(&st16T[col * 136 + kk + grp * 8]),
                         ldh8(&st16T[col * 136 + kk + 16 + grp * 8]));
        acc = wmma16(af, bf, acc);
      }
      #pragma unroll
      for (int r = 0; r < 8; ++r) {
        int c = tm * 16 + grp * 8 + r, j = tn * 16 + m_;
        float wv = (float)w16[((size_t)(s0 + c) * HVH + hv) * DVD + j];
        vnT[j * 72 + c] = (_Float16)(wv - acc[r]);
      }
    }
    __syncthreads();

    // P2: o = (q*e^gc) @ state + Aqk @ vnew  (64x128)
    for (int t = wave; t < 32; t += 8) {
      int tm = t >> 3, tn = t & 7;
      int col = tn * 16 + m_;
      int c = tm * 16 + m_;
      float eg = __expf(gcL[c]);
      const _Float16* qrow = qn + ((size_t)(s0 + c) * HKH + hq) * DKD;
      v8f acc = {};
      #pragma unroll
      for (int kk = 0; kk < DKD; kk += 32) {
        v16h af = scale16(mkfrag(ldh8(&qrow[kk + grp * 8]),
                                 ldh8(&qrow[kk + 16 + grp * 8])), eg);
        v16h bf = mkfrag(ldh8(&st16T[col * 136 + kk + grp * 8]),
                         ldh8(&st16T[col * 136 + kk + 16 + grp * 8]));
        acc = wmma16(af, bf, acc);
      }
      const _Float16* arow = aqk16 + (((size_t)n * HVH + hv) * CHUNK + c) * CHUNK;
      #pragma unroll
      for (int kk = 0; kk < CHUNK; kk += 32) {
        v16h af = mkfrag(ldh8(&arow[kk + grp * 8]), ldh8(&arow[kk + 16 + grp * 8]));
        v16h bf = mkfrag(ldh8(&vnT[col * 72 + kk + grp * 8]),
                         ldh8(&vnT[col * 72 + kk + 16 + grp * 8]));
        acc = wmma16(af, bf, acc);
      }
      #pragma unroll
      for (int r = 0; r < 8; ++r) {
        int cc = tm * 16 + grp * 8 + r, j = tn * 16 + m_;
        core[((size_t)(s0 + cc) * HVH + hv) * DVD + j] = acc[r];
      }
    }
    __syncthreads();

    // P3: state = state*e^glast + k_dec^T @ vnew  (128x128, K=64)
    for (int t = wave; t < 64; t += 8) {
      int tm = t >> 3, tn = t & 7;
      int d = tm * 16 + m_, col = tn * 16 + m_;
      v8f acc = {};
      #pragma unroll
      for (int kk = 0; kk < CHUNK; kk += 32) {
        v16h af = mkfrag(ldh8(&kdT[d * 72 + kk + grp * 8]),
                         ldh8(&kdT[d * 72 + kk + 16 + grp * 8]));
        v16h bf = mkfrag(ldh8(&vnT[col * 72 + kk + grp * 8]),
                         ldh8(&vnT[col * 72 + kk + 16 + grp * 8]));
        acc = wmma16(af, bf, acc);
      }
      #pragma unroll
      for (int r = 0; r < 8; ++r) {
        int dd = tm * 16 + grp * 8 + r, j = tn * 16 + m_;
        float nv = state[dd * 132 + j] * el + acc[r];
        state[dd * 132 + j] = nv;
        st16T[j * 136 + dd] = (_Float16)nv;
      }
    }
    __syncthreads();
  }
}

// ---------------- gated RMSNorm: rmsnorm(core)*norm_w*silu(z) -> f16 --------------
__global__ __launch_bounds__(256) void gdn_gate_norm(
    const float* __restrict__ core, const float* __restrict__ z,
    const float* __restrict__ normw, _Float16* __restrict__ gated)
{
  int row = blockIdx.x * 8 + (threadIdx.x >> 5);
  int lane = threadIdx.x & 31;
  if (row >= SEQ * HVH) return;
  const float* c = core + (size_t)row * DVD;
  float v[4]; float ss = 0.0f;
  #pragma unroll
  for (int i = 0; i < 4; ++i) { v[i] = c[lane + 32 * i]; ss += v[i] * v[i]; }
  ss = wavesum(ss);
  float rs = rsqrtf(ss * (1.0f / DVD) + EPSV);
  const float* zr = z + (size_t)row * DVD;
  _Float16* out = gated + (size_t)row * DVD;
  #pragma unroll
  for (int i = 0; i < 4; ++i) {
    int d = lane + 32 * i;
    out[d] = (_Float16)(v[i] * rs * normw[d] * siluf_(zr[d]));
  }
}

// ---------------- launch ----------------
static inline unsigned cdiv(size_t a, size_t b) { return (unsigned)((a + b - 1) / b); }

extern "C" void kernel_launch(void* const* d_in, const int* in_sizes, int n_in,
                              void* d_out, int out_size, void* d_ws, size_t ws_size,
                              hipStream_t stream) {
  (void)in_sizes; (void)n_in; (void)out_size; (void)ws_size;
  const float* x     = (const float*)d_in[0];
  const float* Wqkv  = (const float*)d_in[1];
  const float* Wz    = (const float*)d_in[2];
  const float* Wb    = (const float*)d_in[3];
  const float* Wa    = (const float*)d_in[4];
  const float* convw = (const float*)d_in[5];
  const float* Alog  = (const float*)d_in[6];
  const float* dtb   = (const float*)d_in[7];
  const float* normw = (const float*)d_in[8];
  const float* Wout  = (const float*)d_in[9];
  float* out = (float*)d_out;

  unsigned char* ws = (unsigned char*)d_ws;
  size_t off = 0;
  auto carve = [&](size_t bytes) -> void* {
    void* p = ws + off;
    off = (off + bytes + 255) & ~(size_t)255;
    return p;
  };

  _Float16* Xh     = (_Float16*)carve((size_t)SEQ * HID * 2);
  _Float16* Wqkvh  = (_Float16*)carve((size_t)HID * CONVD * 2);
  _Float16* Wzh    = (_Float16*)carve((size_t)HID * HVH * DVD * 2);
  _Float16* Wouth  = (_Float16*)carve((size_t)HVH * DVD * HID * 2);
  float*    mixed  = (float*)carve((size_t)SEQ * CONVD * 4);
  float*    zbuf   = (float*)carve((size_t)SEQ * HVH * DVD * 4);
  float*    qraw   = (float*)carve((size_t)SEQ * HKH * DKD * 4);
  float*    kraw   = (float*)carve((size_t)SEQ * HKH * DKD * 4);
  _Float16* qn     = (_Float16*)carve((size_t)SEQ * HKH * DKD * 2);
  _Float16* kn     = (_Float16*)carve((size_t)SEQ * HKH * DKD * 2);
  _Float16* vh     = (_Float16*)carve((size_t)SEQ * HVH * DVD * 2);
  float*    bvec   = (float*)carve((size_t)SEQ * HVH * 4);
  float*    avec   = (float*)carve((size_t)SEQ * HVH * 4);
  float*    beta   = (float*)carve((size_t)SEQ * HVH * 4);
  float*    gbuf   = (float*)carve((size_t)SEQ * HVH * 4);
  float*    gcum   = (float*)carve((size_t)SEQ * HVH * 4);
  _Float16* w16    = (_Float16*)carve((size_t)SEQ * HVH * DVD * 2);
  _Float16* u16    = (_Float16*)carve((size_t)SEQ * HVH * DKD * 2);
  _Float16* aqk16  = (_Float16*)carve((size_t)NCH * HVH * CHUNK * CHUNK * 2);
  float*    coreb  = (float*)carve((size_t)SEQ * HVH * DVD * 4);
  _Float16* gated  = (_Float16*)carve((size_t)SEQ * HVH * DVD * 2);

  // 1. f16 conversions (vectorized)
  gdn_cvt_f16<<<cdiv((size_t)SEQ * HID / 8, 256), 256, 0, stream>>>(x, Xh, (size_t)SEQ * HID);
  gdn_cvt_f16<<<cdiv((size_t)HID * CONVD / 8, 256), 256, 0, stream>>>(Wqkv, Wqkvh, (size_t)HID * CONVD);
  gdn_cvt_f16<<<cdiv((size_t)HID * HVH * DVD / 8, 256), 256, 0, stream>>>(Wz, Wzh, (size_t)HID * HVH * DVD);
  gdn_cvt_f16<<<cdiv((size_t)HVH * DVD * HID / 8, 256), 256, 0, stream>>>(Wout, Wouth, (size_t)HVH * DVD * HID);

  // 2. big projections via WMMA GEMM (async global->LDS staging)
  gdn_gemm_wmma<<<dim3(CONVD / GBN, SEQ / GBM), 256, 0, stream>>>(Xh, Wqkvh, mixed, SEQ, CONVD, HID);
  gdn_gemm_wmma<<<dim3((HVH * DVD) / GBN, SEQ / GBM), 256, 0, stream>>>(Xh, Wzh, zbuf, SEQ, HVH * DVD, HID);

  // 3. small projections + conv + norms + gates
  gdn_proj_ba<<<SEQ / 8, 256, 0, stream>>>(x, Wb, Wa, bvec, avec);
  gdn_conv_silu<<<cdiv((size_t)SEQ * CONVD, 256), 256, 0, stream>>>(mixed, convw, qraw, kraw, vh);
  gdn_l2norm_qk<<<cdiv((size_t)2 * SEQ * HKH, 8), 256, 0, stream>>>(qraw, kraw, qn, kn);
  gdn_gates<<<cdiv((size_t)SEQ * HVH, 256), 256, 0, stream>>>(bvec, avec, Alog, dtb, beta, gbuf);
  gdn_cumsum<<<cdiv((size_t)NCH * HVH, 256), 256, 0, stream>>>(gbuf, gcum);

  // 4. per-chunk WY representation (WMMA + LDS triangular solve)
  gdn_chunk<<<dim3(NCH, HVH), 256, CHK_LDS, stream>>>(qn, kn, vh, beta, gcum, w16, u16, aqk16);

  // 5. sequential inter-chunk scan per head (f32 state + f16 mirror in LDS)
  gdn_scan<<<HVH, 256, SCAN_LDS, stream>>>(qn, kn, w16, u16, aqk16, gcum, coreb);

  // 6. gated RMSNorm + output projection
  gdn_gate_norm<<<cdiv((size_t)SEQ * HVH, 8), 256, 0, stream>>>(coreb, zbuf, normw, gated);
  gdn_gemm_wmma<<<dim3(HID / GBN, SEQ / GBM), 256, 0, stream>>>(gated, Wouth, out, SEQ, HID, HVH * DVD);
}
